// MultiHeadAttention_43447889166739
// MI455X (gfx1250) — compile-verified, matrix operand order checked by reading, not run on hardware
//
#include <hip/hip_runtime.h>

#define BB 4
#define TQn 2048
#define TVn 2048
#define DDn 1024
#define HHn 8
#define DKn 128

typedef __bf16 bf16_t;
typedef __attribute__((ext_vector_type(16))) __bf16 bf16x16;
typedef __attribute__((ext_vector_type(8)))  __bf16 bf16x8;
typedef __attribute__((ext_vector_type(8)))  float  f32x8;

__device__ __forceinline__ bf16x16 load_b16(const bf16_t* p0, const bf16_t* p1) {
    bf16x8 lo = *reinterpret_cast<const bf16x8*>(p0);
    bf16x8 hi = *reinterpret_cast<const bf16x8*>(p1);
    return __builtin_shufflevector(lo, hi, 0,1,2,3,4,5,6,7,8,9,10,11,12,13,14,15);
}

// A-fragment from fp32 source: elements [0..7] and [16..23] relative to p.
__device__ __forceinline__ bf16x16 cvt_a_f32(const float* p) {
    bf16x16 r;
#pragma unroll
    for (int i = 0; i < 8; ++i) { r[i] = (bf16_t)p[i]; r[8 + i] = (bf16_t)p[16 + i]; }
    return r;
}

__device__ __forceinline__ f32x8 wmma_bf16(bf16x16 a, bf16x16 b, f32x8 c) {
    return __builtin_amdgcn_wmma_f32_16x16x32_bf16(false, a, false, b, (short)0, c, false, false);
}

// CDNA5 async DMA: global -> LDS, 16 bytes per lane, tracked by ASYNCcnt.
__device__ __forceinline__ void async_ld16(unsigned lds_off, const void* gptr) {
    asm volatile("global_load_async_to_lds_b128 %0, %1, off"
                 :: "v"(lds_off), "v"((unsigned long long)(uintptr_t)gptr)
                 : "memory");
}

// -------- weight convert + transpose:  src[batch][rows][cols] f32 -> dst[batch][cols][rows] bf16
__global__ void __launch_bounds__(256)
wconv_kernel(const float* __restrict__ src, bf16_t* __restrict__ dst, int rows, int cols) {
    long long idx = (long long)blockIdx.x * 256 + threadIdx.x;
    long long per = (long long)rows * cols;
    long long bidx = idx / per;
    long long rem  = idx - bidx * per;
    int r = (int)(rem / cols);
    int c = (int)(rem - (long long)r * cols);
    dst[bidx * per + (long long)c * rows + r] = (bf16_t)src[idx];
}

// -------- per-head projection: x[b*T+t][D] f32 × wt[h][DK][D] bf16 -> outr[bh][T][DK] (+ optional outt[bh][DK][T])
__global__ void __launch_bounds__(256)
proj_kernel(const float* __restrict__ x, const bf16_t* __restrict__ wt,
            bf16_t* __restrict__ outr, bf16_t* __restrict__ outt, int T) {
    const int lane = threadIdx.x & 31;
    const int wid  = threadIdx.x >> 5;
    const int tile = blockIdx.x * 8 + wid;
    const int tiles_per_bh = T >> 4;
    const int bh = tile / tiles_per_bh;
    const int r0 = (tile % tiles_per_bh) << 4;
    const int b  = bh / HHn;
    const int h  = bh % HHn;
    const int nlo   = lane & 15;
    const int kb    = (lane & 16) ? 8 : 0;
    const int rbase = (lane & 16) ? 8 : 0;

    const float* xrow = x + (long long)(b * T + r0 + nlo) * DDn;
    f32x8 acc[8] = {};

    for (int d0 = 0; d0 < DDn; d0 += 32) {
        bf16x16 a = cvt_a_f32(xrow + d0 + kb);
#pragma unroll
        for (int j = 0; j < 8; ++j) {
            const bf16_t* w = wt + (long long)(h * DKn + j * 16 + nlo) * DDn + d0 + kb;
            bf16x16 bfrag = load_b16(w, w + 16);
            acc[j] = wmma_bf16(a, bfrag, acc[j]);
        }
    }
#pragma unroll
    for (int j = 0; j < 8; ++j) {
#pragma unroll
        for (int r = 0; r < 8; ++r) {
            int row = r0 + rbase + r;
            int col = j * 16 + nlo;
            bf16_t v = (bf16_t)acc[j][r];
            outr[(long long)bh * T * DKn + (long long)row * DKn + col] = v;
            if (outt) outt[(long long)bh * DKn * T + (long long)col * T + row] = v;
        }
    }
}

// -------- flash attention --------
// Block = 8 waves, one (b,h) and 128 q rows. Per 32-key step the block stages
// K-tile (32x128) and Vt-tile (128x32) into LDS via async DMA (double-buffered),
// then each wave runs 8 WMMAs for scores, online softmax, and 8 WMMAs for P*V.
__global__ void __launch_bounds__(256)
attn_kernel(const bf16_t* __restrict__ qb, const bf16_t* __restrict__ vb,
            const bf16_t* __restrict__ vt, const unsigned char* __restrict__ qmask,
            const unsigned char* __restrict__ vmask, const float* __restrict__ scale,
            bf16_t* __restrict__ ob) {
    __shared__ bf16_t sK[2][32 * DKn];   // [buf][key_row][dk]   8KB each
    __shared__ bf16_t sV[2][DKn * 32];   // [buf][dk][key_row]   8KB each
    __shared__ bf16_t sP[8][512];        // per-wave 16x32 P restripe tile

    const int tid  = threadIdx.x;
    const int lane = tid & 31;
    const int wid  = tid >> 5;
    const int bh   = blockIdx.x >> 4;          // TQ/128 = 16 q-chunks per (b,h)
    const int b    = bh / HHn;
    const int h    = bh % HHn;
    const int q0   = ((blockIdx.x & 15) << 7) + (wid << 4);
    const int nlo   = lane & 15;
    const int kb    = (lane & 16) ? 8 : 0;
    const int rbase = (lane & 16) ? 8 : 0;
    bf16_t* P = &sP[wid][0];

    const bf16_t* vbBase = vb + (long long)bh * TVn * DKn;   // row-major [TV][DK]
    const bf16_t* vtBase = vt + (long long)bh * DKn * TVn;   // row-major [DK][TV]

    // cooperative async stage of one 32-key tile into buffer `buf`
    auto stage = [&](int buf, int t0) {
#pragma unroll
        for (int p = 0; p < 2; ++p) {
            int idx = tid + p * 256;                         // 0..511, 16B each
            // K tile: contiguous 8KB starting at row t0
            async_ld16((unsigned)(uintptr_t)&sK[buf][0] + (unsigned)idx * 16,
                       (const char*)(vbBase + (long long)t0 * DKn) + (long long)idx * 16);
            // Vt tile: 128 rows x 64B, row stride TV*2 bytes
            int r = idx >> 2, qq = idx & 3;
            async_ld16((unsigned)(uintptr_t)&sV[buf][0] + (unsigned)idx * 16,
                       (const char*)(vtBase + (long long)r * TVn + t0) + qq * 16);
        }
    };

    // Q fragments for this wave's 16 rows (DK = 4 chunks of K=32)
    bf16x16 qa[4];
    const bf16_t* qrow = qb + (long long)(bh * TQn + q0 + nlo) * DKn;
#pragma unroll
    for (int kk = 0; kk < 4; ++kk)
        qa[kk] = load_b16(qrow + kk * 32 + kb, qrow + kk * 32 + kb + 16);

    const float sc = scale[h];
    f32x8 O[8] = {};
    float mrun[8], srun[8];
#pragma unroll
    for (int r = 0; r < 8; ++r) { mrun[r] = -3.0e38f; srun[r] = 0.0f; }

    stage(0, 0);   // prologue: tile 0 in flight

    for (int t0 = 0; t0 < TVn; t0 += 32) {
        const int buf = (t0 >> 5) & 1;
        if (t0 + 32 < TVn) {
            stage(buf ^ 1, t0 + 32);                       // prefetch next tile
            asm volatile("s_wait_asynccnt 4" ::: "memory"); // current tile's 4 ops done
        } else {
            asm volatile("s_wait_asynccnt 0" ::: "memory");
        }
        __syncthreads();   // all waves' DMA landed

        // scores S = Q * K^T  (two 16x16 C tiles covering 32 keys) from LDS
        f32x8 c0 = {}, c1 = {};
#pragma unroll
        for (int kk = 0; kk < 4; ++kk) {
            const bf16_t* vr0 = &sK[buf][nlo * DKn + kk * 32 + kb];
            const bf16_t* vr1 = vr0 + 16 * DKn;
            bf16x16 b0 = load_b16(vr0, vr0 + 16);
            bf16x16 b1 = load_b16(vr1, vr1 + 16);
            c0 = wmma_bf16(qa[kk], b0, c0);
            c1 = wmma_bf16(qa[kk], b1, c1);
        }
        const bool mk0 = vmask[b * TVn + t0 + nlo] != 0;
        const bool mk1 = vmask[b * TVn + t0 + 16 + nlo] != 0;
        float alpha[8];
#pragma unroll
        for (int r = 0; r < 8; ++r) {
            float s0 = mk0 ? c0[r] * sc : -1.0e9f;
            float s1 = mk1 ? c1[r] * sc : -1.0e9f;
            float mt = fmaxf(s0, s1);                 // row-wise max within 16-lane half
            mt = fmaxf(mt, __shfl_xor(mt, 1));
            mt = fmaxf(mt, __shfl_xor(mt, 2));
            mt = fmaxf(mt, __shfl_xor(mt, 4));
            mt = fmaxf(mt, __shfl_xor(mt, 8));
            float mn = fmaxf(mrun[r], mt);
            float al = __expf(mrun[r] - mn);
            mrun[r] = mn;
            float e0 = __expf(s0 - mn);
            float e1 = __expf(s1 - mn);
            float rs = e0 + e1;
            rs += __shfl_xor(rs, 1);
            rs += __shfl_xor(rs, 2);
            rs += __shfl_xor(rs, 4);
            rs += __shfl_xor(rs, 8);
            srun[r] = srun[r] * al + rs;
            alpha[r] = al;
            c0[r] = e0; c1[r] = e1;
        }
#pragma unroll
        for (int j = 0; j < 8; ++j)
#pragma unroll
            for (int r = 0; r < 8; ++r) O[j][r] *= alpha[r];

        // restripe P: C-layout -> row-major LDS -> A-layout (LDS in-order per wave)
#pragma unroll
        for (int r = 0; r < 8; ++r) {
            P[(rbase + r) * 32 + nlo]      = (bf16_t)c0[r];
            P[(rbase + r) * 32 + 16 + nlo] = (bf16_t)c1[r];
        }
        asm volatile("s_wait_dscnt 0" ::: "memory");
        bf16x16 pa = load_b16(P + nlo * 32 + kb, P + nlo * 32 + kb + 16);
        asm volatile("" ::: "memory");

        // O += P * V  (Vt tile in LDS, row-major [dk][key])
#pragma unroll
        for (int j = 0; j < 8; ++j) {
            const bf16_t* vtp = &sV[buf][(j * 16 + nlo) * 32 + kb];
            bf16x16 bfrag = load_b16(vtp, vtp + 16);
            O[j] = wmma_bf16(pa, bfrag, O[j]);
        }
        __syncthreads();   // everyone done reading buf before it is re-staged
    }

    // normalize, apply q_mask, store heads-concatenated bf16 [B][TQ][H*DK]
#pragma unroll
    for (int r = 0; r < 8; ++r) {
        int row = q0 + rbase + r;
        float inv = 1.0f / srun[r];
        float qm = qmask[b * TQn + row] ? inv : 0.0f;
#pragma unroll
        for (int j = 0; j < 8; ++j)
            ob[((long long)b * TQn + row) * (HHn * DKn) + h * DKn + j * 16 + nlo]
                = (bf16_t)(O[j][r] * qm);
    }
}

// -------- output projection: [8192,1024] bf16 × wot[1024(n)][1024(k)] bf16 -> f32 out
__global__ void __launch_bounds__(256)
oproj_kernel(const bf16_t* __restrict__ hm, const bf16_t* __restrict__ wot,
             float* __restrict__ out) {
    const int lane = threadIdx.x & 31;
    const int wid  = threadIdx.x >> 5;
    const int rt = blockIdx.x >> 3;
    const int ct = blockIdx.x & 7;
    const int r0 = rt * 128 + wid * 16;
    const int c0 = ct * 128;
    const int nlo   = lane & 15;
    const int kb    = (lane & 16) ? 8 : 0;
    const int rbase = (lane & 16) ? 8 : 0;

    const bf16_t* arow = hm + (long long)(r0 + nlo) * DDn;
    f32x8 acc[8] = {};
    for (int k0 = 0; k0 < DDn; k0 += 32) {
        bf16x16 a = load_b16(arow + k0 + kb, arow + k0 + kb + 16);
#pragma unroll
        for (int j = 0; j < 8; ++j) {
            const bf16_t* w = wot + (long long)(c0 + j * 16 + nlo) * DDn + k0 + kb;
            bf16x16 bfrag = load_b16(w, w + 16);
            acc[j] = wmma_bf16(a, bfrag, acc[j]);
        }
    }
#pragma unroll
    for (int j = 0; j < 8; ++j)
#pragma unroll
        for (int r = 0; r < 8; ++r)
            out[(long long)(r0 + rbase + r) * DDn + c0 + j * 16 + nlo] = acc[j][r];
}

extern "C" void kernel_launch(void* const* d_in, const int* in_sizes, int n_in,
                              void* d_out, int out_size, void* d_ws, size_t ws_size,
                              hipStream_t stream) {
    (void)in_sizes; (void)n_in; (void)out_size; (void)ws_size;
    const float*         query  = (const float*)d_in[0];
    const float*         value  = (const float*)d_in[1];
    const unsigned char* q_mask = (const unsigned char*)d_in[2];
    const unsigned char* v_mask = (const unsigned char*)d_in[3];
    const float*         qw     = (const float*)d_in[4];
    const float*         vw     = (const float*)d_in[5];
    const float*         wo     = (const float*)d_in[6];
    const float*         scale  = (const float*)d_in[7];
    float*               out    = (float*)d_out;

    char* ws = (char*)d_ws;
    size_t off = 0;
    auto alloc = [&](size_t bytes) -> void* {
        void* p = ws + off;
        off += (bytes + 255) & ~(size_t)255;
        return p;
    };
    bf16_t* qwt = (bf16_t*)alloc((size_t)HHn * DKn * DDn * 2);      // [H][DK][D]
    bf16_t* vwt = (bf16_t*)alloc((size_t)HHn * DKn * DDn * 2);      // [H][DK][D]
    bf16_t* wot = (bf16_t*)alloc((size_t)DDn * HHn * DKn * 2);      // [D(n)][H*DK(k)]
    bf16_t* qb  = (bf16_t*)alloc((size_t)BB * HHn * TQn * DKn * 2); // [bh][TQ][DK]
    bf16_t* vb  = (bf16_t*)alloc((size_t)BB * HHn * TVn * DKn * 2); // [bh][TV][DK]
    bf16_t* vtb = (bf16_t*)alloc((size_t)BB * HHn * DKn * TVn * 2); // [bh][DK][TV]
    bf16_t* ob  = (bf16_t*)alloc((size_t)BB * TQn * HHn * DKn * 2); // [B][TQ][H*DK]

    // 1. weight transforms (f32 -> bf16, K innermost)
    wconv_kernel<<<(HHn * DDn * DKn) / 256, 256, 0, stream>>>(qw, qwt, DDn, DKn);
    wconv_kernel<<<(HHn * DDn * DKn) / 256, 256, 0, stream>>>(vw, vwt, DDn, DKn);
    wconv_kernel<<<(HHn * DKn * DDn) / 256, 256, 0, stream>>>(wo, wot, HHn * DKn, DDn);

    // 2. projections (V also written transposed for the attn*V B-fragments)
    proj_kernel<<<BB * HHn * (TQn / 16) / 8, 256, 0, stream>>>(query, qwt, qb, nullptr, TQn);
    proj_kernel<<<BB * HHn * (TVn / 16) / 8, 256, 0, stream>>>(value, vwt, vb, vtb, TVn);

    // 3. flash attention (LDS-staged, async double-buffered)
    attn_kernel<<<BB * HHn * (TQn / 128), 256, 0, stream>>>(qb, vb, vtb, q_mask, v_mask, scale, ob);

    // 4. output projection
    oproj_kernel<<<(BB * TQn / 128) * (DDn / 128), 256, 0, stream>>>(ob, wot, out);
}
